// fpn_classifier_graph_71339406787085
// MI455X (gfx1250) — matrix operand order repaired and checked by reading, not compiled, not run on hardware
//
#include <hip/hip_runtime.h>
#include <hip/hip_bf16.h>

// ---------------------------------------------------------------------------
// Types for CDNA5 WMMA
// ---------------------------------------------------------------------------
typedef __attribute__((ext_vector_type(16))) __bf16 v16bf;
typedef __attribute__((ext_vector_type(8)))  float  v8f;
typedef unsigned short u16;

// Problem constants (from reference)
#define NB    2
#define NROI  1000
#define CCH   256
#define FDIM  1024
#define MV    2000          // valid rows
#define MP    2048          // padded rows (multiple of 128)
#define K1    12544         // 7*7*256
#define POOLP 7
#define EPSF  0.001f

// GEMM tiling
#define BLK_M     128
#define BLK_N     128
#define KSTEP     64                    // K per LDS stage (2 wmma sub-steps)
#define LDS_ROWB  144                   // padded row pitch (bytes) for 128B rows
#define TILE_B    (128 * LDS_ROWB)      // 18432 B per tile
#define STAGE_B   (2 * TILE_B)          // A + B tiles = 36864 B
#define SMEM_B    (2 * STAGE_B)         // double buffered = 73728 B

__device__ __forceinline__ u16 f2bf(float f) {
    union { float f; unsigned u; } v; v.f = f;
    unsigned r = v.u + 0x7FFFu + ((v.u >> 16) & 1u);   // round-to-nearest-even
    return (u16)(r >> 16);
}

// ---------------------------------------------------------------------------
// CDNA5 async global->LDS copy (ASYNCcnt-tracked), per ISA ch.10 / 08_async
// ---------------------------------------------------------------------------
__device__ __forceinline__ void async_b128(unsigned long long gaddr, unsigned ldsOff) {
    asm volatile("global_load_async_to_lds_b128 %0, %1, off"
                 :: "v"(ldsOff), "v"(gaddr) : "memory");
}
__device__ __forceinline__ void wait_async_le8() {
    asm volatile("s_wait_asynccnt 0x8" ::: "memory");
}
__device__ __forceinline__ void wait_async_le0() {
    asm volatile("s_wait_asynccnt 0x0" ::: "memory");
}

// ---------------------------------------------------------------------------
// 1) Weight transpose + f32->bf16:  in[k*F + f]  ->  out[f*K + k]
// ---------------------------------------------------------------------------
__global__ void transpose_to_bf16(const float* __restrict__ in, u16* __restrict__ out,
                                  int K, int F) {
    __shared__ float tile[32][33];
    int kb = blockIdx.x * 32;
    int fb = blockIdx.y * 32;
    int tx = threadIdx.x;      // 0..31
    int ty = threadIdx.y;      // 0..7
#pragma unroll
    for (int i = 0; i < 4; ++i) {
        int kr = ty + i * 8;
        tile[kr][tx] = in[(size_t)(kb + kr) * F + fb + tx];
    }
    __syncthreads();
#pragma unroll
    for (int i = 0; i < 4; ++i) {
        int fr = ty + i * 8;
        out[(size_t)(fb + fr) * K + kb + tx] = f2bf(tile[tx][fr]);
    }
}

// ---------------------------------------------------------------------------
// 2) Pyramid ROI align -> bf16 A-matrix [MP][K1], rows >= MV zero-padded.
// ---------------------------------------------------------------------------
__global__ void roi_align_bf16(const float* __restrict__ rois,
                               const float* __restrict__ p2, const float* __restrict__ p3,
                               const float* __restrict__ p4, const float* __restrict__ p5,
                               u16* __restrict__ A1) {
    int pos = blockIdx.x;              // 0 .. MP*49-1
    int m   = pos / 49;
    int pp  = pos % 49;
    int ph  = pp / POOLP, pw = pp % POOLP;
    int c   = threadIdx.x;             // 0..255
    size_t outIdx = (size_t)m * K1 + (size_t)pp * CCH + c;
    if (m >= MV) { A1[outIdx] = 0; return; }

    int b = m / NROI;
    float y1 = rois[m * 4 + 0], x1 = rois[m * 4 + 1];
    float y2 = rois[m * 4 + 2], x2 = rois[m * 4 + 3];
    float h = y2 - y1, w = x2 - x1;
    // level selection: scale = 256/sqrt(1024*1024) = 0.25
    float lvl = log2f(fmaxf(sqrtf(h * w), 1e-9f) * 4.0f);
    int li = (int)rintf(lvl) + 4;
    li = min(max(li, 2), 5);
    const float* fm; int H;
    if      (li == 2) { fm = p2; H = 256; }
    else if (li == 3) { fm = p3; H = 128; }
    else if (li == 4) { fm = p4; H = 64;  }
    else              { fm = p5; H = 32;  }

    const float inv6 = 1.0f / 6.0f;
    float ys = (y1 + h * (float)ph * inv6) * (float)(H - 1);
    float xs = (x1 + w * (float)pw * inv6) * (float)(H - 1);
    float y0f = floorf(ys), x0f = floorf(xs);
    float wy = ys - y0f,  wx = xs - x0f;
    int y0i = min(max((int)y0f, 0), H - 1);
    int y1i = min(y0i + 1, H - 1);
    int x0i = min(max((int)x0f, 0), H - 1);
    int x1i = min(x0i + 1, H - 1);

    size_t base = (size_t)b * H * H * CCH;
    float v00 = fm[base + ((size_t)y0i * H + x0i) * CCH + c];
    float v01 = fm[base + ((size_t)y0i * H + x1i) * CCH + c];
    float v10 = fm[base + ((size_t)y1i * H + x0i) * CCH + c];
    float v11 = fm[base + ((size_t)y1i * H + x1i) * CCH + c];
    float top = v00 + (v01 - v00) * wx;
    float bot = v10 + (v11 - v10) * wx;
    A1[outIdx] = f2bf(top + (bot - top) * wy);
}

// ---------------------------------------------------------------------------
// 3) Workgroup-cooperative WMMA bf16 GEMM with async global->LDS staging.
//    C[M][N] = A[M][K] * BT[N][K]^T + bias[N]
//    Block: 256 threads (8 waves) -> 128x128 tile; wave -> 64(M) x 32(N).
//    Per 64-K stage: A(128x64 bf16) + B(128x64 bf16) staged in LDS,
//    double-buffered, copied with global_load_async_to_lds_b128.
// ---------------------------------------------------------------------------
__device__ __forceinline__ v16bf lds_a_frag(const char* bufA, int lane, int mLocal, int kk) {
    // lanes 0-15: M=mLocal+lane,   K offsets {0..7,16..23}
    // lanes16-31: M=mLocal+lane-16,K offsets {8..15,24..31}   (+kk*32)
    const char* p = bufA + (size_t)(mLocal + (lane & 15)) * LDS_ROWB
                        + kk * 64 + ((lane >> 4) << 4);
    union { uint4 q[2]; v16bf v; } u;
    u.q[0] = *(const uint4*)(p);        // K + {0..7}
    u.q[1] = *(const uint4*)(p + 32);   // K + {16..23}
    return u.v;
}

__device__ __forceinline__ v16bf lds_b_frag(const char* bufB, int lane, int nLocal, int kk) {
    // lanes 0-15: N=nLocal+lane, K = {0..7,16..23} ; lanes 16-31: K = {8..15,24..31}  (+kk*32)
    const char* p = bufB + (size_t)(nLocal + (lane & 15)) * LDS_ROWB
                        + kk * 64 + ((lane >> 4) << 4);
    union { uint4 q[2]; v16bf v; } u;
    u.q[0] = *(const uint4*)(p);
    u.q[1] = *(const uint4*)(p + 32);
    return u.v;
}

__global__ void gemm_bf16_wmma(const u16* __restrict__ A, const u16* __restrict__ BT,
                               const float* __restrict__ bias, float* __restrict__ C,
                               int M, int N, int K) {
    extern __shared__ char smem[];
    const int tid  = threadIdx.x;
    const int lane = tid & 31;
    const int wave = tid >> 5;

    const int nBlkTiles = N / BLK_N;
    const int mBlock = (blockIdx.x / nBlkTiles) * BLK_M;
    const int nBlock = (blockIdx.x % nBlkTiles) * BLK_N;

    const int waveM = (wave & 1) * 64;       // 0 or 64
    const int waveN = (wave >> 1) * 32;      // 0,32,64,96

    // copy-slice coordinates: each thread moves one 16B chunk per issue
    const int cRow = tid >> 3;               // 0..31
    const int cCol = (tid & 7) * 16;         // byte col 0..112 (row = 128B)

    const int KT = K / KSTEP;

    // stage-copy lambda: 8 async b128 issues per wave per stage
    auto stage_copy = [&](int s) {
        char* bufA = smem + (s & 1) * STAGE_B;
        char* bufB = bufA + TILE_B;
        const char* Ab = (const char*)A + (size_t)s * KSTEP * 2;   // + row*K*2 below
        const char* Bb = (const char*)BT + (size_t)s * KSTEP * 2;
#pragma unroll
        for (int i = 0; i < 4; ++i) {
            int r = cRow + i * 32;           // 0..127
            async_b128((unsigned long long)(Ab + (size_t)(mBlock + r) * K * 2 + cCol),
                       (unsigned)(size_t)(bufA + r * LDS_ROWB + cCol));
            async_b128((unsigned long long)(Bb + (size_t)(nBlock + r) * K * 2 + cCol),
                       (unsigned)(size_t)(bufB + r * LDS_ROWB + cCol));
        }
    };

    v8f acc[4][2] = {};

    stage_copy(0);
    for (int s = 0; s < KT; ++s) {
        if (s + 1 < KT) { stage_copy(s + 1); wait_async_le8(); }
        else            { wait_async_le0(); }
        __syncthreads();                     // stage-s tile valid for all waves

        const char* bufA = smem + (s & 1) * STAGE_B;
        const char* bufB = bufA + TILE_B;
#pragma unroll
        for (int kk = 0; kk < 2; ++kk) {     // two 32-K wmma sub-steps
            v16bf b0 = lds_b_frag(bufB, lane, waveN,      kk);
            v16bf b1 = lds_b_frag(bufB, lane, waveN + 16, kk);
#pragma unroll
            for (int i = 0; i < 4; ++i) {
                v16bf a = lds_a_frag(bufA, lane, waveM + 16 * i, kk);
                acc[i][0] = __builtin_amdgcn_wmma_f32_16x16x32_bf16(
                    false, a, false, b0, (short)0, acc[i][0], false, false);
                acc[i][1] = __builtin_amdgcn_wmma_f32_16x16x32_bf16(
                    false, a, false, b1, (short)0, acc[i][1], false, false);
            }
        }
        __syncthreads();                     // all reads done before overwrite
    }

    // C/D layout: VGPR r -> M = r (+8 for lanes 16-31), N = lane & 15
    const int halfsel = lane >> 4;
    const int col     = lane & 15;
#pragma unroll
    for (int i = 0; i < 4; ++i) {
#pragma unroll
        for (int j = 0; j < 2; ++j) {
            int cIdx = nBlock + waveN + 16 * j + col;
            float bv = bias[cIdx];
#pragma unroll
            for (int r = 0; r < 8; ++r) {
                int row = mBlock + waveM + 16 * i + r + halfsel * 8;
                C[(size_t)row * N + cIdx] = acc[i][j][r] + bv;
            }
        }
    }
}

// ---------------------------------------------------------------------------
// 4) Batch-norm (training stats over MV rows), 3-phase
// ---------------------------------------------------------------------------
#define BN_CHUNKS 16

__global__ void bn_partial(const float* __restrict__ x, float* __restrict__ psum,
                           float* __restrict__ psumsq) {
    int fblk  = blockIdx.x % (FDIM / 256);
    int chunk = blockIdx.x / (FDIM / 256);
    int f = fblk * 256 + threadIdx.x;
    int per = (MV + BN_CHUNKS - 1) / BN_CHUNKS;           // 125
    int m0 = chunk * per;
    int m1 = min(MV, m0 + per);
    float s = 0.f, sq = 0.f;
    for (int m = m0; m < m1; ++m) {
        float v = x[(size_t)m * FDIM + f];
        s += v; sq += v * v;
    }
    psum[chunk * FDIM + f]   = s;
    psumsq[chunk * FDIM + f] = sq;
}

__global__ void bn_finalize(const float* __restrict__ psum, const float* __restrict__ psumsq,
                            const float* __restrict__ gamma, const float* __restrict__ beta,
                            float* __restrict__ scale, float* __restrict__ shift) {
    int f = blockIdx.x * 256 + threadIdx.x;
    float s = 0.f, sq = 0.f;
    for (int c = 0; c < BN_CHUNKS; ++c) { s += psum[c * FDIM + f]; sq += psumsq[c * FDIM + f]; }
    const float invM = 1.0f / (float)MV;
    float mean = s * invM;
    float var  = sq * invM - mean * mean;
    float sc   = gamma[f] * rsqrtf(var + EPSF);
    scale[f] = sc;
    shift[f] = beta[f] - mean * sc;
}

__global__ void bn_apply(const float* __restrict__ x, const float* __restrict__ scale,
                         const float* __restrict__ shift, u16* __restrict__ outBf,
                         float* __restrict__ outF) {
    size_t i = (size_t)blockIdx.x * blockDim.x + threadIdx.x;   // MP*FDIM elements
    int m = (int)(i / FDIM);
    int f = (int)(i % FDIM);
    float v = 0.f;
    if (m < MV) v = fmaxf(x[i] * scale[f] + shift[f], 0.f);
    if (outBf) outBf[i] = f2bf(v);
    if (outF)  outF[i]  = v;
}

// ---------------------------------------------------------------------------
// 5) Heads: logits / softmax probs / deltas.  One block per ROI.
//    d_out layout: logits[2000*2] | probs[2000*2] | deltas[2000*8]
// ---------------------------------------------------------------------------
__global__ void heads_kernel(const float* __restrict__ S,
                             const float* __restrict__ cls_w, const float* __restrict__ cls_b,
                             const float* __restrict__ dlt_w, const float* __restrict__ dlt_b,
                             float* __restrict__ out) {
    int m = blockIdx.x;                 // 0..MV-1
    int t = threadIdx.x;                // 0..255
    const float* row = S + (size_t)m * FDIM;
    float acc[10];
#pragma unroll
    for (int j = 0; j < 10; ++j) acc[j] = 0.f;
    for (int k = t; k < FDIM; k += 256) {
        float s = row[k];
        acc[0] += s * cls_w[k * 2 + 0];
        acc[1] += s * cls_w[k * 2 + 1];
#pragma unroll
        for (int j = 0; j < 8; ++j) acc[2 + j] += s * dlt_w[k * 8 + j];
    }
    __shared__ float red[256];
    __shared__ float res[10];
    for (int j = 0; j < 10; ++j) {
        red[t] = acc[j];
        __syncthreads();
        for (int o = 128; o > 0; o >>= 1) {
            if (t < o) red[t] += red[t + o];
            __syncthreads();
        }
        if (t == 0) res[j] = red[0];
        __syncthreads();
    }
    if (t == 0) {
        float l0 = res[0] + cls_b[0];
        float l1 = res[1] + cls_b[1];
        out[m * 2 + 0] = l0;
        out[m * 2 + 1] = l1;
        float mx = fmaxf(l0, l1);
        float e0 = expf(l0 - mx), e1 = expf(l1 - mx);
        float inv = 1.f / (e0 + e1);
        out[2 * MV + m * 2 + 0] = e0 * inv;
        out[2 * MV + m * 2 + 1] = e1 * inv;
#pragma unroll
        for (int j = 0; j < 8; ++j)
            out[4 * MV + m * 8 + j] = res[2 + j] + dlt_b[j];
    }
}

// ---------------------------------------------------------------------------
// Host launcher
// ---------------------------------------------------------------------------
extern "C" void kernel_launch(void* const* d_in, const int* in_sizes, int n_in,
                              void* d_out, int out_size, void* d_ws, size_t ws_size,
                              hipStream_t stream) {
    const float* rois    = (const float*)d_in[0];
    const float* p2      = (const float*)d_in[1];
    const float* p3      = (const float*)d_in[2];
    const float* p4      = (const float*)d_in[3];
    const float* p5      = (const float*)d_in[4];
    const float* conv1_w = (const float*)d_in[5];
    const float* conv1_b = (const float*)d_in[6];
    const float* bn1_g   = (const float*)d_in[7];
    const float* bn1_b   = (const float*)d_in[8];
    const float* conv2_w = (const float*)d_in[9];
    const float* conv2_b = (const float*)d_in[10];
    const float* bn2_g   = (const float*)d_in[11];
    const float* bn2_b   = (const float*)d_in[12];
    const float* cls_w   = (const float*)d_in[13];
    const float* cls_b   = (const float*)d_in[14];
    const float* dlt_w   = (const float*)d_in[15];
    const float* dlt_b   = (const float*)d_in[16];
    float* out = (float*)d_out;

    // Workspace layout (all 256B aligned)
    char* ws = (char*)d_ws;
    size_t off = 0;
    auto alloc = [&](size_t bytes) { char* p = ws + off; off += (bytes + 255) & ~size_t(255); return p; };
    u16*   A1    = (u16*)  alloc((size_t)MP * K1 * 2);        // 51.4 MB  pooled bf16
    u16*   W1T   = (u16*)  alloc((size_t)FDIM * K1 * 2);      // 25.7 MB
    u16*   W2T   = (u16*)  alloc((size_t)FDIM * FDIM * 2);    //  2.1 MB
    float* X1    = (float*)alloc((size_t)MP * FDIM * 4);      //  8.4 MB
    u16*   A2    = (u16*)  alloc((size_t)MP * FDIM * 2);      //  4.2 MB
    float* X2    = (float*)alloc((size_t)MP * FDIM * 4);      //  8.4 MB
    float* S2    = (float*)alloc((size_t)MP * FDIM * 4);      //  8.4 MB
    float* psum  = (float*)alloc((size_t)BN_CHUNKS * FDIM * 4);
    float* psumq = (float*)alloc((size_t)BN_CHUNKS * FDIM * 4);
    float* scale = (float*)alloc((size_t)FDIM * 4);
    float* shift = (float*)alloc((size_t)FDIM * 4);
    (void)ws_size; (void)in_sizes; (void)n_in; (void)out_size;

    // 1) weights -> bf16 transposed [N][K]
    transpose_to_bf16<<<dim3(K1 / 32, FDIM / 32), dim3(32, 8), 0, stream>>>(conv1_w, W1T, K1, FDIM);
    transpose_to_bf16<<<dim3(FDIM / 32, FDIM / 32), dim3(32, 8), 0, stream>>>(conv2_w, W2T, FDIM, FDIM);

    // 2) ROI align -> bf16 A-matrix
    roi_align_bf16<<<MP * 49, CCH, 0, stream>>>(rois, p2, p3, p4, p5, A1);

    // 3) GEMM1 (2048 x 1024 x 12544) -> X1
    {
        int blocks = (MP / BLK_M) * (FDIM / BLK_N);   // 16 * 8 = 128
        gemm_bf16_wmma<<<blocks, 256, SMEM_B, stream>>>(A1, W1T, conv1_b, X1, MP, FDIM, K1);
    }

    // 4) BN1 + ReLU -> A2 (bf16)
    bn_partial<<<BN_CHUNKS * (FDIM / 256), 256, 0, stream>>>(X1, psum, psumq);
    bn_finalize<<<FDIM / 256, 256, 0, stream>>>(psum, psumq, bn1_g, bn1_b, scale, shift);
    bn_apply<<<(MP * FDIM) / 256, 256, 0, stream>>>(X1, scale, shift, A2, nullptr);

    // 5) GEMM2 (2048 x 1024 x 1024) -> X2
    {
        int blocks = (MP / BLK_M) * (FDIM / BLK_N);
        gemm_bf16_wmma<<<blocks, 256, SMEM_B, stream>>>(A2, W2T, conv2_b, X2, MP, FDIM, FDIM);
    }

    // 6) BN2 + ReLU -> S2 (f32)
    bn_partial<<<BN_CHUNKS * (FDIM / 256), 256, 0, stream>>>(X2, psum, psumq);
    bn_finalize<<<FDIM / 256, 256, 0, stream>>>(psum, psumq, bn2_g, bn2_b, scale, shift);
    bn_apply<<<(MP * FDIM) / 256, 256, 0, stream>>>(X2, scale, shift, nullptr, S2);

    // 7) Heads -> d_out
    heads_kernel<<<MV, 256, 0, stream>>>(S2, cls_w, cls_b, dlt_w, dlt_b, out);
}
